// VAE_15324443312225
// MI455X (gfx1250) — compile-verified, matrix operand order checked by reading, not run on hardware
//
#include <hip/hip_runtime.h>
#include <hip/hip_bf16.h>
#include <type_traits>

// ---------------- problem dims ----------------
#define BATCH 4096
#define IMG   27648
#define HDIM  300
#define HP    320     // H padded to 16-multiple (20 N-tiles), zero-padded
#define NODE  4

typedef __attribute__((ext_vector_type(8)))  float  v8f;
typedef __attribute__((ext_vector_type(8)))  __bf16 v8bf;
typedef __attribute__((ext_vector_type(16))) __bf16 v16bf;
typedef int          v4i_  __attribute__((vector_size(16)));
typedef unsigned int u32x4 __attribute__((vector_size(16)));
typedef int          i32x8 __attribute__((vector_size(32)));
typedef int          i32x4 __attribute__((vector_size(16)));

#ifndef __has_builtin
#define __has_builtin(x) 0
#endif
#if __has_builtin(__builtin_amdgcn_global_load_async_to_lds_b128)
#define HAVE_ASYNC 1
#else
#define HAVE_ASYNC 0
#endif
#if __has_builtin(__builtin_amdgcn_s_wait_asynccnt)
#define HAVE_WAIT_ASYNC 1
#else
#define HAVE_WAIT_ASYNC 0
#endif
#if __has_builtin(__builtin_amdgcn_tensor_load_to_lds) && \
    __has_builtin(__builtin_amdgcn_s_wait_tensorcnt)
#define HAVE_TDM 1
#else
#define HAVE_TDM 0
#endif

__device__ __forceinline__ float elu1(float x) { return x > 0.f ? x : __expf(x) - 1.f; }

__device__ __forceinline__ v16bf cat16(v8bf lo, v8bf hi) {
  return __builtin_shufflevector(lo, hi, 0,1,2,3,4,5,6,7,8,9,10,11,12,13,14,15);
}

// 16B global -> LDS copy: async on CDNA5 (ASYNCcnt), sync fallback otherwise.
__device__ __forceinline__ void cp16(const void* g, void* l) {
#if HAVE_ASYNC
  __builtin_amdgcn_global_load_async_to_lds_b128(
      (__attribute__((address_space(1))) v4i_*)const_cast<void*>(g),
      (__attribute__((address_space(3))) v4i_*)l, 0, 0);
#else
  *(uint4*)l = *(const uint4*)g;
#endif
}

#if HAVE_TDM
// Tensor Data Mover: 2D tile [tile_d1 rows x tile_d0 elems] from a row-major
// tensor with row stride stride_d0 (elems) -> LDS with pad_amount DWORDs of
// padding after each (1<<pad_interval)*2 DWORDs (i.e. per row here).
// D# layout per CDNA5 ISA ch.8: group0 = {flags, lds_addr, global_addr, type},
// group1 = {dsize/pad, tensor dims, tile dims, dim0 stride}, groups 2-3 = 0.
__device__ __forceinline__ void tdm_load_2d(
    const void* gaddr, void* laddr, unsigned tile_d0, unsigned tile_d1,
    unsigned stride_d0, int dsize_code, int pad_interval, int pad_amount)
{
  unsigned long long ga = (unsigned long long)gaddr;
  unsigned lds = (unsigned)(unsigned long long)
      (__attribute__((address_space(3))) void*)laddr;
  u32x4 g0 = {0u, 0u, 0u, 0u};
  g0[0] = 1u;                                            // count=1, user D#
  g0[1] = lds;                                           // lds_addr
  g0[2] = (unsigned)(ga & 0xffffffffu);                  // global_addr[31:0]
  g0[3] = (unsigned)((ga >> 32) & 0x1ffffffu) | (2u << 30);  // [56:32] | type=2
  unsigned td0 = tile_d0, td1 = tile_d1;                 // tensor dims = tile (in-bounds)
  i32x8 g1 = {0, 0, 0, 0, 0, 0, 0, 0};
  g1[0] = (int)(((unsigned)dsize_code << 16) | (1u << 20) |
                ((unsigned)pad_interval << 22) | ((unsigned)pad_amount << 25));
  g1[1] = (int)((td0 & 0xffffu) << 16);                  // tensor_dim0[15:0]
  g1[2] = (int)(((td0 >> 16) & 0xffffu) | ((td1 & 0xffffu) << 16));
  g1[3] = (int)(((td1 >> 16) & 0xffffu) | ((tile_d0 & 0xffffu) << 16));
  g1[4] = (int)(tile_d1 & 0xffffu);                      // tile_dim1, tile_dim2=0
  g1[5] = (int)stride_d0;                                // dim0_stride[31:0]
  g1[6] = 0;                                             // stride[47:32]=0, d1s=0
  g1[7] = 0;
  i32x4 z4 = {0, 0, 0, 0};
  i32x8 z8 = {0, 0, 0, 0, 0, 0, 0, 0};
  __builtin_amdgcn_tensor_load_to_lds(g0, g1, z4, z4, z8, 0);
}
#endif

// ---------------------------------------------------------------------------
// Weight convert + transpose + pad:
//   src  : f32 [Ksrc][Nsrc]  (row-major, K = reduction dim)
//   dst  : bf16 [Npad][Kpad] (N-major so GEMM B-fragments are K-contiguous)
// ---------------------------------------------------------------------------
__global__ __launch_bounds__(256) void convert_wt_kernel(
    const float* __restrict__ src, __bf16* __restrict__ dst,
    int Ksrc, int Nsrc, int Kpad, int Npad)
{
  long total = (long)Kpad * (long)Npad;
  for (long i = (long)blockIdx.x * 256 + threadIdx.x; i < total;
       i += (long)gridDim.x * 256) {
    int n = (int)(i / Kpad);
    int k = (int)(i % Kpad);
    float v = (k < Ksrc && n < Nsrc) ? src[(long)k * Nsrc + n] : 0.f;
    dst[i] = (__bf16)v;
  }
}

// ---------------------------------------------------------------------------
// WMMA GEMM:  out = act( A[M x K] @ Bt^T + bias )
//   A  : f32 (A_F32=1, staged raw, converted to bf16 at fragment build) or bf16
//   Bt : bf16 [N][ldbt] (pre-transposed weights, K contiguous per row)
//   Block tile 128 x (NT*16), 256 threads = 8 waves; wave w owns rows
//   [w*16, w*16+16) x all NT column tiles.  K-step 32.
//   Double-buffered LDS; staging via TDM (tensor_load_to_lds + TENSORcnt) when
//   available, else per-lane async global->LDS (ASYNCcnt), else sync fallback.
//   LDS rows padded (80B bf16 / 144B f32) for conflict-free ds_load_b128; the
//   TDM path reproduces the same layout via D# pad_enable.
//   ACT=0: ELU -> bf16 store ; ACT=1: tanh -> f32 store.
// ---------------------------------------------------------------------------
template<bool A_F32, int NT, int ACT>
__global__ __launch_bounds__(256) void gemm_wmma_kernel(
    const void* __restrict__ Av, const __bf16* __restrict__ Bt,
    const float* __restrict__ bias, void* __restrict__ outv,
    int K, int lda, int ldbt, int nvalid, int ldout)
{
  constexpr int BN    = NT * 16;
  constexpr int ESZ   = A_F32 ? 4 : 2;      // A element bytes
  constexpr int ACH   = A_F32 ? 8 : 4;      // 16B chunks per A row (32 elems)
  constexpr int SAROW = A_F32 ? 144 : 80;   // padded LDS A row bytes
  constexpr int SBROW = 80;                 // padded LDS B row bytes
  constexpr int ASZ   = 128 * SAROW;
  constexpr int BSZ   = BN * SBROW;
  constexpr int ACNT  = (128 * ACH) / 256;  // async issues per thread (A)
  constexpr int BCNT  = (BN * 4) / 256;     // async issues per thread (B)
  constexpr int NIF   = HAVE_TDM ? 2 : (ACNT + BCNT);  // in-flight per stage()

  __shared__ __align__(16) unsigned char smem[2 * ASZ + 2 * BSZ];

  const int tid  = threadIdx.x;
  const int wave = tid >> 5;
  const int lane = tid & 31;
  const int m0   = blockIdx.x * 128;
  const int n0   = blockIdx.y * BN;

  v8f acc[NT];
  #pragma unroll
  for (int t = 0; t < NT; ++t) acc[t] = v8f{0.f,0.f,0.f,0.f,0.f,0.f,0.f,0.f};

  auto stage = [&](int buf, int k0) {
    unsigned char* aDst = smem + buf * ASZ;
    unsigned char* bDst = smem + 2 * ASZ + buf * BSZ;
#if HAVE_TDM
    if (tid < 32) {   // one wave drives the TDM; 2 descriptors per stage
      tdm_load_2d((const unsigned char*)Av + ((size_t)m0 * lda + k0) * ESZ,
                  aDst, 32, 128, (unsigned)lda,
                  A_F32 ? 2 : 1, A_F32 ? 4 : 3, 3);
      tdm_load_2d((const unsigned char*)Bt + ((size_t)n0 * ldbt + k0) * 2,
                  bDst, 32, BN, (unsigned)ldbt, 1, 3, 3);
    }
#else
    #pragma unroll
    for (int i = 0; i < ACNT; ++i) {
      int g = tid + 256 * i;
      int row = g / ACH, ch = g % ACH;
      const unsigned char* gs = (const unsigned char*)Av +
          ((size_t)(m0 + row) * lda + k0) * ESZ + ch * 16;
      cp16(gs, aDst + row * SAROW + ch * 16);
    }
    #pragma unroll
    for (int i = 0; i < BCNT; ++i) {
      int g = tid + 256 * i;
      int row = g >> 2, ch = g & 3;
      const unsigned char* gs = (const unsigned char*)Bt +
          ((size_t)(n0 + row) * ldbt + k0) * 2 + ch * 16;
      cp16(gs, bDst + row * SBROW + ch * 16);
    }
#endif
  };

  auto wait_stage = [&](auto NC) {   // NC: integral_constant immediate
#if HAVE_TDM
    if (tid < 32) __builtin_amdgcn_s_wait_tensorcnt(NC.value);
#elif HAVE_WAIT_ASYNC
    __builtin_amdgcn_s_wait_asynccnt(NC.value);
#elif HAVE_ASYNC
    asm volatile("s_wait_asynccnt %0" :: "i"((int)NC.value));
#endif
    (void)NC;
  };

  stage(0, 0);
  int buf = 0;
  for (int k0 = 0; k0 < K; k0 += 32, buf ^= 1) {
    // issue next tile into other buffer, then wait for current tile only
    if (k0 + 32 < K) {
      stage(buf ^ 1, k0 + 32);
      wait_stage(std::integral_constant<int, NIF>{});
    } else {
      wait_stage(std::integral_constant<int, 0>{});
    }
    __syncthreads();

    const unsigned char* aCur = smem + buf * ASZ;
    const unsigned char* bCur = smem + 2 * ASZ + buf * BSZ;

    // 16-bit A layout: lane&15 = M row; lane>>4 selects K chunks {0..7,16..23}
    // vs {8..15,24..31}.  B mirrors with lane&15 = N column.
    const int lrow  = lane & 15;
    const int lhalf = lane >> 4;

    v16bf afrag;
    if (A_F32) {
      const float* pa =
          (const float*)(aCur + (wave * 16 + lrow) * SAROW + lhalf * 32);
      float4 x0 = ((const float4*)pa)[0];
      float4 x1 = ((const float4*)pa)[1];
      const float* pa2 = (const float*)((const unsigned char*)pa + 64);
      float4 x2 = ((const float4*)pa2)[0];
      float4 x3 = ((const float4*)pa2)[1];
      afrag[ 0]=(__bf16)x0.x; afrag[ 1]=(__bf16)x0.y; afrag[ 2]=(__bf16)x0.z; afrag[ 3]=(__bf16)x0.w;
      afrag[ 4]=(__bf16)x1.x; afrag[ 5]=(__bf16)x1.y; afrag[ 6]=(__bf16)x1.z; afrag[ 7]=(__bf16)x1.w;
      afrag[ 8]=(__bf16)x2.x; afrag[ 9]=(__bf16)x2.y; afrag[10]=(__bf16)x2.z; afrag[11]=(__bf16)x2.w;
      afrag[12]=(__bf16)x3.x; afrag[13]=(__bf16)x3.y; afrag[14]=(__bf16)x3.z; afrag[15]=(__bf16)x3.w;
    } else {
      const unsigned char* pa = aCur + (wave * 16 + lrow) * SAROW + lhalf * 16;
      v8bf alo = *(const v8bf*)pa;
      v8bf ahi = *(const v8bf*)(pa + 32);
      afrag = cat16(alo, ahi);
    }

    // 2-deep rotated B pipeline: loads for nt+2 in flight while wmma nt runs,
    // so the pre-WMMA wait is a partial s_wait_dscnt, not 0.
    const unsigned char* pbBase = bCur + lrow * SBROW + lhalf * 16;
    v8bf blo0 = *(const v8bf*)pbBase;
    v8bf bhi0 = *(const v8bf*)(pbBase + 32);
    v8bf blo1 = *(const v8bf*)(pbBase + 16 * SBROW);
    v8bf bhi1 = *(const v8bf*)(pbBase + 16 * SBROW + 32);
    #pragma unroll
    for (int nt = 0; nt < NT; ++nt) {
      v8bf nlo = blo1, nhi = bhi1;
      if (nt + 2 < NT) {
        const unsigned char* pb = pbBase + (nt + 2) * 16 * SBROW;
        nlo = *(const v8bf*)pb;
        nhi = *(const v8bf*)(pb + 32);
      }
      acc[nt] = __builtin_amdgcn_wmma_f32_16x16x32_bf16(
          false, afrag, false, cat16(blo0, bhi0), (short)0, acc[nt], false, false);
      blo0 = blo1; bhi0 = bhi1; blo1 = nlo; bhi1 = nhi;
    }
    __syncthreads();   // all waves done with buf before it is restaged
  }

  // ---- epilogue: bias + activation + store ----
  // C/D layout: col = lane&15 (+n-tile), VGPR r -> row r (lanes 0-15) / 8+r.
  const int erow0 = m0 + wave * 16 + (lane >> 4) * 8;
  const int ecol  = lane & 15;
  #pragma unroll
  for (int nt = 0; nt < NT; ++nt) {
    int colg = n0 + nt * 16 + ecol;
    float b = (colg < nvalid) ? bias[colg] : 0.f;
    #pragma unroll
    for (int r = 0; r < 8; ++r) {
      int rowg = erow0 + r;
      float v = acc[nt][r] + b;
      if (ACT == 0) {
        v = elu1(v);  // padded cols stay exactly 0 through ELU
        ((__bf16*)outv)[(size_t)rowg * ldout + colg] = (__bf16)v;
      } else {
        ((float*)outv)[(size_t)rowg * ldout + colg] = tanhf(v);
      }
    }
  }
}

// ---------------------------------------------------------------------------
// encoder L3 (+ reparameterized noise + prior): tiny N=4 GEMM, VALU.
// ---------------------------------------------------------------------------
__global__ __launch_bounds__(256) void enc3_reparam_prior_kernel(
    const __bf16* __restrict__ h2, const float* __restrict__ W3,
    const float* __restrict__ b3, const float* __restrict__ noise,
    const float* __restrict__ label, const float* __restrict__ pw,
    const float* __restrict__ pb, float* __restrict__ out_logvar,
    float* __restrict__ out_prior, float* __restrict__ eps)
{
  int i = blockIdx.x * 256 + threadIdx.x;
  if (i >= BATCH * NODE) return;
  int row = i >> 2, j = i & 3;
  const __bf16* hr = h2 + (size_t)row * HP;
  float s = b3[j];
  for (int k = 0; k < HDIM; ++k) s += (float)hr[k] * W3[k * NODE + j];
  out_logvar[i] = s;
  eps[i] = __expf(0.5f * s) * noise[i];
  out_prior[i] = label[i] * pw[j] + pb[j];
}

// ---------------------------------------------------------------------------
// Sequential SCM: per-row chain over 4 nodes, 5->4->2->1 MLP, ELU/tanh.
// ---------------------------------------------------------------------------
__global__ __launch_bounds__(256) void scm_kernel(
    const float* __restrict__ eps, const float* __restrict__ Bm,
    const float* __restrict__ W1, const float* __restrict__ b1,
    const float* __restrict__ W2, const float* __restrict__ b2,
    const float* __restrict__ W3, const float* __restrict__ b3,
    float* __restrict__ out_latent, float* __restrict__ out_align,
    float* __restrict__ lat_ws)
{
  int row = blockIdx.x * 256 + threadIdx.x;
  if (row >= BATCH) return;
  float lat[NODE] = {0.f, 0.f, 0.f, 0.f};
  float e[NODE];
  #pragma unroll
  for (int j = 0; j < NODE; ++j) e[j] = eps[row * NODE + j];

  #pragma unroll
  for (int j = 0; j < NODE; ++j) {
    float z[NODE + 1];
    #pragma unroll
    for (int p = 0; p < NODE; ++p) z[p] = lat[p] * Bm[p * NODE + j];
    z[NODE] = e[j];
    float g1[4];
    #pragma unroll
    for (int o = 0; o < 4; ++o) {
      float s = b1[j * 4 + o];
      #pragma unroll
      for (int p = 0; p < NODE + 1; ++p) s += z[p] * W1[j * 20 + p * 4 + o];
      g1[o] = elu1(s);
    }
    float g2[2];
    #pragma unroll
    for (int o = 0; o < 2; ++o) {
      float s = b2[j * 2 + o];
      #pragma unroll
      for (int p = 0; p < 4; ++p) s += g1[p] * W2[j * 8 + p * 2 + o];
      g2[o] = elu1(s);
    }
    float a = b3[j] + g2[0] * W3[j * 2 + 0] + g2[1] * W3[j * 2 + 1];
    out_align[row * NODE + j] = a;
    lat[j] = tanhf(a);
  }
  #pragma unroll
  for (int j = 0; j < NODE; ++j) {
    out_latent[row * NODE + j] = lat[j];
    lat_ws[row * NODE + j] = lat[j];
  }
}

// ---------------------------------------------------------------------------
// decoder L1: hd1 = ELU(latent[4096x4] @ dec_W1[4x300] + b1) -> bf16 [4096xHP]
// ---------------------------------------------------------------------------
__global__ __launch_bounds__(256) void dec1_kernel(
    const float* __restrict__ lat, const float* __restrict__ W1d,
    const float* __restrict__ b1d, __bf16* __restrict__ hd1)
{
  int i = blockIdx.x * 256 + threadIdx.x;
  if (i >= BATCH * HP) return;
  int row = i / HP, c = i % HP;
  float v = 0.f;
  if (c < HDIM) {
    v = b1d[c];
    #pragma unroll
    for (int j = 0; j < NODE; ++j) v += lat[row * NODE + j] * W1d[j * HDIM + c];
    v = elu1(v);
  }
  hd1[i] = (__bf16)v;
}

// ---------------------------------------------------------------------------
extern "C" void kernel_launch(void* const* d_in, const int* in_sizes, int n_in,
                              void* d_out, int out_size, void* d_ws, size_t ws_size,
                              hipStream_t stream) {
  const float* image   = (const float*)d_in[0];
  const float* label   = (const float*)d_in[1];
  const float* noise   = (const float*)d_in[2];
  const float* Bm      = (const float*)d_in[3];
  const float* enc_W1  = (const float*)d_in[4];
  const float* enc_b1  = (const float*)d_in[5];
  const float* enc_W2  = (const float*)d_in[6];
  const float* enc_b2  = (const float*)d_in[7];
  const float* enc_W3  = (const float*)d_in[8];
  const float* enc_b3  = (const float*)d_in[9];
  const float* nW1     = (const float*)d_in[10];
  const float* nb1     = (const float*)d_in[11];
  const float* nW2     = (const float*)d_in[12];
  const float* nb2     = (const float*)d_in[13];
  const float* nW3     = (const float*)d_in[14];
  const float* nb3     = (const float*)d_in[15];
  const float* dec_W1  = (const float*)d_in[16];
  const float* dec_b1  = (const float*)d_in[17];
  const float* dec_W2  = (const float*)d_in[18];
  const float* dec_b2  = (const float*)d_in[19];
  const float* dec_W3  = (const float*)d_in[20];
  const float* dec_b3  = (const float*)d_in[21];
  const float* prior_w = (const float*)d_in[22];
  const float* prior_b = (const float*)d_in[23];

  // outputs, concatenated flat: logvar | prior_logvar | latent | align | xhat
  float* out       = (float*)d_out;
  float* o_logvar  = out;
  float* o_prior   = out + (size_t)BATCH * NODE;
  float* o_latent  = out + (size_t)BATCH * NODE * 2;
  float* o_align   = out + (size_t)BATCH * NODE * 3;
  float* o_xhat    = out + (size_t)BATCH * NODE * 4;

  // workspace layout (bytes, 256-aligned)
  char* ws = (char*)d_ws;
  constexpr size_t SZ_W1T  = (size_t)HP  * IMG * 2;
  constexpr size_t SZ_W3T  = (size_t)IMG * HP  * 2;
  constexpr size_t SZ_W2T  = (size_t)HP  * HP  * 2;
  constexpr size_t SZ_H    = (size_t)BATCH * HP * 2;
  constexpr size_t SZ_EPS  = (size_t)BATCH * NODE * 4;

  size_t off = 0;
  __bf16* W1T  = (__bf16*)(ws + off); off += SZ_W1T;
  __bf16* W3DT = (__bf16*)(ws + off); off += SZ_W3T;
  __bf16* W2ET = (__bf16*)(ws + off); off += SZ_W2T;
  __bf16* W2DT = (__bf16*)(ws + off); off += SZ_W2T;
  __bf16* H1   = (__bf16*)(ws + off); off += SZ_H;
  __bf16* H2E  = (__bf16*)(ws + off); off += SZ_H;
  __bf16* HD1  = (__bf16*)(ws + off); off += SZ_H;
  __bf16* HD2  = (__bf16*)(ws + off); off += SZ_H;
  float*  EPS  = (float*)(ws + off);  off += SZ_EPS;
  float*  LAT  = (float*)(ws + off);  off += SZ_EPS;
  (void)off; (void)in_sizes; (void)n_in; (void)out_size; (void)ws_size;

  // 1) weight conversions (f32 [K][N] -> bf16 [Npad][Kpad], transposed+padded)
  convert_wt_kernel<<<8192, 256, 0, stream>>>(enc_W1, W1T,  IMG,  HDIM, IMG, HP);
  convert_wt_kernel<<< 512, 256, 0, stream>>>(enc_W2, W2ET, HDIM, HDIM, HP,  HP);
  convert_wt_kernel<<< 512, 256, 0, stream>>>(dec_W2, W2DT, HDIM, HDIM, HP,  HP);
  convert_wt_kernel<<<8192, 256, 0, stream>>>(dec_W3, W3DT, HDIM, IMG,  HP,  IMG);

  // 2) encoder L1: h1 = ELU(image @ W1 + b1)   [4096 x HP] bf16
  gemm_wmma_kernel<true, 20, 0><<<dim3(BATCH / 128, 1), 256, 0, stream>>>(
      (const void*)image, W1T, enc_b1, (void*)H1, IMG, IMG, IMG, HDIM, HP);

  // 3) encoder L2: h2 = ELU(h1 @ W2 + b2)
  gemm_wmma_kernel<false, 20, 0><<<dim3(BATCH / 128, 1), 256, 0, stream>>>(
      (const void*)H1, W2ET, enc_b2, (void*)H2E, HP, HP, HP, HDIM, HP);

  // 4) encoder L3 + reparameterize + prior
  enc3_reparam_prior_kernel<<<(BATCH * NODE) / 256, 256, 0, stream>>>(
      H2E, enc_W3, enc_b3, noise, label, prior_w, prior_b,
      o_logvar, o_prior, EPS);

  // 5) sequential SCM over nodes
  scm_kernel<<<BATCH / 256, 256, 0, stream>>>(
      EPS, Bm, nW1, nb1, nW2, nb2, nW3, nb3, o_latent, o_align, LAT);

  // 6) decoder L1 (K=4, VALU)
  dec1_kernel<<<(BATCH * HP) / 256, 256, 0, stream>>>(LAT, dec_W1, dec_b1, HD1);

  // 7) decoder L2: hd2 = ELU(hd1 @ dec_W2 + b2)
  gemm_wmma_kernel<false, 20, 0><<<dim3(BATCH / 128, 1), 256, 0, stream>>>(
      (const void*)HD1, W2DT, dec_b2, (void*)HD2, HP, HP, HP, HDIM, HP);

  // 8) decoder L3: xhat = tanh(hd2 @ dec_W3 + b3)   [4096 x 27648] f32
  gemm_wmma_kernel<false, 16, 1><<<dim3(BATCH / 128, IMG / 256), 256, 0, stream>>>(
      (const void*)HD2, W3DT, dec_b3, (void*)o_xhat, HP, HP, HP, IMG, IMG);
}
